// MessagePassingLayer_43001212567950
// MI455X (gfx1250) — matrix operand order repaired and checked by reading, not compiled, not run on hardware
//
#include <hip/hip_runtime.h>

// ---------- types ----------
typedef __attribute__((ext_vector_type(16))) __bf16        v16bf;
typedef __attribute__((ext_vector_type(8)))  float         v8f;
typedef __attribute__((ext_vector_type(4)))  unsigned int  u32x4;
typedef unsigned short u16;

union FragBF {
    v16bf bf;
    u32x4 u[2];
};

#define NODE_F 64
#define EDGE_F 32
#define OUT_F  64
#define HIDDEN 128
#define IN_F   160           // 2*NODE_F + EDGE_F
#define EPB    128           // edges per block
#define A_STRIDE 168         // IN_F padded (bf16 elems) -> 84 dwords/row, conflict-free
#define H_STRIDE 136         // HIDDEN padded (bf16 elems) -> 68 dwords/row

__device__ __forceinline__ u16 f2bf(float f) {
    unsigned u = __float_as_uint(f);
    u += 0x7FFFu + ((u >> 16) & 1u);   // round-to-nearest-even
    return (u16)(u >> 16);
}

__device__ __forceinline__ v8f wmma_bf16(FragBF a, FragBF b, v8f c) {
    // D = A(16x32 bf16) * B(32x16 bf16) + C(16x16 f32)
    return __builtin_amdgcn_wmma_f32_16x16x32_bf16(
        false, a.bf, false, b.bf, (short)0, c, false, false);
}

// A-fragment: lane holds row (l&15); two 16B chunks: K = hi*8+{0..7}, 16+hi*8+{0..7}
__device__ __forceinline__ FragBF load_fragA(const u16* rowPtr, int kb, int hi) {
    FragBF a;
    const u16* p = rowPtr + kb + hi * 8;
    a.u[0] = *(const u32x4*)(p);
    a.u[1] = *(const u32x4*)(p + 16);
    return a;
}

// B-fragment from transposed weights WT[n][k] (k contiguous, length Ktot):
// lane holds column (l&15); two 16B chunks: K = hi*8+{0..7}, 16+hi*8+{0..7}
__device__ __forceinline__ FragBF load_fragB(const u16* __restrict__ WT,
                                             int n, int Ktot, int kb, int hi) {
    FragBF b;
    const u16* p = WT + n * Ktot + kb + hi * 8;
    b.u[0] = *(const u32x4*)(p);
    b.u[1] = *(const u32x4*)(p + 16);
    return b;
}

// ---------- prep kernels ----------
__global__ void k_f32_to_bf16(const float* __restrict__ src, u16* __restrict__ dst, int n) {
    int i = blockIdx.x * 256 + threadIdx.x;
    if (i < n) dst[i] = f2bf(src[i]);
}

// W: [K][N] row-major f32  ->  WT: [N][K] bf16
__global__ void k_convert_transpose(const float* __restrict__ W, u16* __restrict__ WT,
                                    int K, int N) {
    int i = blockIdx.x * 256 + threadIdx.x;
    if (i < K * N) {
        int k = i / N, n = i % N;
        WT[n * K + k] = f2bf(W[i]);
    }
}

__global__ void k_degree(const int* __restrict__ recv, float* __restrict__ deg, int E) {
    int i = blockIdx.x * 256 + threadIdx.x;
    if (i < E) unsafeAtomicAdd(&deg[recv[i]], 1.0f);
}

// ---------- edge MLP: 128 edges per block, 8 waves ----------
__global__ __launch_bounds__(256)
void k_edge_mlp(const float* __restrict__ edges,
                const int*   __restrict__ senders,
                const int*   __restrict__ receivers,
                const u16*   __restrict__ nodesbf,   // [n_nodes][64] bf16
                const u16*   __restrict__ W1T,       // [128][160] bf16
                const u16*   __restrict__ W2T,       // [64][128] bf16
                const float* __restrict__ b1,
                const float* __restrict__ b2,
                float*       __restrict__ agg,
                int E) {
    // sA (128 x A_STRIDE bf16 = 43008 B) aliased with sH (128 x H_STRIDE = 34816 B);
    // sA is dead after the layer-1 K-loop (barrier-protected reuse).
    __shared__ alignas(16) u16 smem[EPB * A_STRIDE];
    __shared__ int sRecv[EPB];
    u16* sA = smem;
    u16* sH = smem;

    const int tid   = threadIdx.x;
    const int ebase = blockIdx.x * EPB;

    if (tid < EPB) {
        int e = ebase + tid;
        sRecv[tid] = (e < E) ? receivers[e] : -1;
    }

    // Stage concat tile: 128 edges x 160 bf16 = 128 x 80 dwords (zero-pad tail edges)
    const unsigned* nodes_u = (const unsigned*)nodesbf;   // 2 bf16 per dword, 32/row
    for (int i = tid; i < EPB * 80; i += 256) {
        int le = i / 80, p = i % 80;
        int e = ebase + le;
        unsigned v = 0u;
        if (e < E) {
            if (p < 32) {
                v = nodes_u[senders[e] * 32 + p];
            } else if (p < 64) {
                v = nodes_u[receivers[e] * 32 + (p - 32)];
            } else {
                int f = (p - 64) * 2;
                float f0 = edges[e * EDGE_F + f];
                float f1 = edges[e * EDGE_F + f + 1];
                v = (unsigned)f2bf(f0) | ((unsigned)f2bf(f1) << 16);
            }
        }
        ((unsigned*)sA)[le * (A_STRIDE / 2) + p] = v;
    }
    __syncthreads();

    const int wave = tid >> 5, lane = tid & 31;
    const int hi = lane >> 4, lr = lane & 15;
    const int mp = wave & 3;        // M-pair: M-tiles 2*mp, 2*mp+1 (32 edges)
    const int ng = wave >> 2;       // column half: 0 -> N 0..63, 1 -> N 64..127

    // ---- layer 1: [128x160] @ [160x128]; each B fragment feeds 2 WMMAs ----
    v8f acc[2][4] = {};             // zero accumulators; bias folded into epilogue
    const u16* aRow0 = &sA[(mp * 32 + lr) * A_STRIDE];
    const u16* aRow1 = aRow0 + 16 * A_STRIDE;
#pragma unroll
    for (int ks = 0; ks < 5; ks++) {
        FragBF a0 = load_fragA(aRow0, ks * 32, hi);
        FragBF a1 = load_fragA(aRow1, ks * 32, hi);
#pragma unroll
        for (int nt = 0; nt < 4; nt++) {
            FragBF b = load_fragB(W1T, ng * 64 + nt * 16 + lr, IN_F, ks * 32, hi);
            acc[0][nt] = wmma_bf16(a0, b, acc[0][nt]);
            acc[1][nt] = wmma_bf16(a1, b, acc[1][nt]);
        }
    }
    __syncthreads();   // all layer-1 reads of sA complete before sH overwrites it

    // bias + ReLU -> bf16 -> sH  (C layout: lane col = lr, vgpr r row = r + 8*hi)
#pragma unroll
    for (int nt = 0; nt < 4; nt++) {
        int n = ng * 64 + nt * 16 + lr;
        float bv = b1[n];
#pragma unroll
        for (int i = 0; i < 2; i++) {
            int rowBase = mp * 32 + i * 16 + 8 * hi;
#pragma unroll
            for (int r = 0; r < 8; r++) {
                float v = acc[i][nt][r] + bv;
                v = v > 0.0f ? v : 0.0f;
                sH[(rowBase + r) * H_STRIDE + n] = f2bf(v);
            }
        }
    }
    __syncthreads();

    // ---- layer 2: [128x128] @ [128x64]; 2 M-tiles x 2 N-tiles per wave ----
    v8f acc2[2][2] = {};
    const u16* hRow0 = &sH[(mp * 32 + lr) * H_STRIDE];
    const u16* hRow1 = hRow0 + 16 * H_STRIDE;
#pragma unroll
    for (int ks = 0; ks < 4; ks++) {
        FragBF a0 = load_fragA(hRow0, ks * 32, hi);
        FragBF a1 = load_fragA(hRow1, ks * 32, hi);
#pragma unroll
        for (int j = 0; j < 2; j++) {
            FragBF b = load_fragB(W2T, (ng * 2 + j) * 16 + lr, HIDDEN, ks * 32, hi);
            acc2[0][j] = wmma_bf16(a0, b, acc2[0][j]);
            acc2[1][j] = wmma_bf16(a1, b, acc2[1][j]);
        }
    }
    // bias + scatter messages (agg is L2-resident; native f32 atomic add)
#pragma unroll
    for (int j = 0; j < 2; j++) {
        int n = (ng * 2 + j) * 16 + lr;
        float bv = b2[n];
#pragma unroll
        for (int i = 0; i < 2; i++) {
            int rowBase = mp * 32 + i * 16 + 8 * hi;
#pragma unroll
            for (int r = 0; r < 8; r++) {
                int rv = sRecv[rowBase + r];
                if (rv >= 0) unsafeAtomicAdd(&agg[rv * OUT_F + n], acc2[i][j][r] + bv);
            }
        }
    }
}

// ---------- finalize: out = agg/max(deg,1) + nodes@Wn + bn ----------
__global__ __launch_bounds__(256)
void k_finalize(const float* __restrict__ agg,
                const float* __restrict__ deg,
                const u16*   __restrict__ nodesbf,
                const u16*   __restrict__ WnT,      // [64][64] bf16
                const float* __restrict__ bn,
                float*       __restrict__ out,
                int n_nodes) {
    const int wave = threadIdx.x >> 5, lane = threadIdx.x & 31;
    const int hi = lane >> 4, lr = lane & 15;
    const int mbase = blockIdx.x * 128 + wave * 16;

    v8f acc[4] = {};                 // zero accumulators; bias in epilogue
    int arow = mbase + lr;
    if (arow >= n_nodes) arow = 0;   // clamp, keep EXEC all-ones for WMMA
    const u16* rowPtr = nodesbf + arow * NODE_F;
#pragma unroll
    for (int ks = 0; ks < 2; ks++) {
        FragBF a = load_fragA(rowPtr, ks * 32, hi);
#pragma unroll
        for (int nt = 0; nt < 4; nt++) {
            FragBF b = load_fragB(WnT, nt * 16 + lr, NODE_F, ks * 32, hi);
            acc[nt] = wmma_bf16(a, b, acc[nt]);
        }
    }
#pragma unroll
    for (int r = 0; r < 8; r++) {
        int row = mbase + r + 8 * hi;
        if (row < n_nodes) {
            float d = deg[row];
            float rd = 1.0f / (d < 1.0f ? 1.0f : d);
#pragma unroll
            for (int nt = 0; nt < 4; nt++) {
                int n = nt * 16 + lr;
                out[row * OUT_F + n] = agg[row * OUT_F + n] * rd + acc[nt][r] + bn[n];
            }
        }
    }
}

// ---------- host ----------
extern "C" void kernel_launch(void* const* d_in, const int* in_sizes, int n_in,
                              void* d_out, int out_size, void* d_ws, size_t ws_size,
                              hipStream_t stream) {
    const float* nodes     = (const float*)d_in[0];
    const float* edges     = (const float*)d_in[1];
    const int*   senders   = (const int*)d_in[2];
    const int*   receivers = (const int*)d_in[3];
    const float* W1        = (const float*)d_in[4];
    const float* b1        = (const float*)d_in[5];
    const float* W2        = (const float*)d_in[6];
    const float* b2        = (const float*)d_in[7];
    const float* Wn        = (const float*)d_in[8];
    const float* bn        = (const float*)d_in[9];
    float* out = (float*)d_out;

    const int n_nodes = in_sizes[0] / NODE_F;
    const int E       = in_sizes[2];

    char* ws = (char*)d_ws;
    size_t off = 0;
    auto alloc = [&](size_t bytes) -> void* {
        void* p = ws + off;
        off = (off + bytes + 255) & ~(size_t)255;
        return p;
    };
    float* agg     = (float*)alloc((size_t)n_nodes * OUT_F * sizeof(float));
    float* deg     = (float*)alloc((size_t)n_nodes * sizeof(float));
    u16*   nodesbf = (u16*)  alloc((size_t)n_nodes * NODE_F * sizeof(u16));
    u16*   W1T     = (u16*)  alloc((size_t)IN_F * HIDDEN * sizeof(u16));
    u16*   W2T     = (u16*)  alloc((size_t)HIDDEN * OUT_F * sizeof(u16));
    u16*   WnT     = (u16*)  alloc((size_t)NODE_F * OUT_F * sizeof(u16));
    (void)ws_size; (void)n_in; (void)out_size;

    hipMemsetAsync(agg, 0, (size_t)n_nodes * OUT_F * sizeof(float), stream);
    hipMemsetAsync(deg, 0, (size_t)n_nodes * sizeof(float), stream);

    int nElemNodes = n_nodes * NODE_F;
    k_f32_to_bf16<<<(nElemNodes + 255) / 256, 256, 0, stream>>>(nodes, nodesbf, nElemNodes);
    k_convert_transpose<<<(IN_F * HIDDEN + 255) / 256, 256, 0, stream>>>(W1, W1T, IN_F, HIDDEN);
    k_convert_transpose<<<(HIDDEN * OUT_F + 255) / 256, 256, 0, stream>>>(W2, W2T, HIDDEN, OUT_F);
    k_convert_transpose<<<(NODE_F * OUT_F + 255) / 256, 256, 0, stream>>>(Wn, WnT, NODE_F, OUT_F);
    k_degree<<<(E + 255) / 256, 256, 0, stream>>>(receivers, deg, E);

    k_edge_mlp<<<(E + EPB - 1) / EPB, 256, 0, stream>>>(edges, senders, receivers, nodesbf,
                                                        W1T, W2T, b1, b2, agg, E);
    k_finalize<<<(n_nodes + 127) / 128, 256, 0, stream>>>(agg, deg, nodesbf, WnT, bn,
                                                          out, n_nodes);
}
